// CoreGroupConstruction_24610162606763
// MI455X (gfx1250) — matrix operand order repaired and checked by reading, not compiled, not run on hardware
//
#include <hip/hip_runtime.h>
#include <hip/hip_bf16.h>

typedef __attribute__((ext_vector_type(16))) __bf16 v16bf;
typedef __attribute__((ext_vector_type(8)))  float  v8f;
typedef __attribute__((ext_vector_type(4)))  unsigned int v4u;
typedef __attribute__((ext_vector_type(8)))  int  v8i;
typedef __attribute__((ext_vector_type(4)))  int  v4i;

#define M_DIM  8192
#define NC_DIM 2048
#define K_DIM  32

// TDM availability (device pass only; host pass falls back cleanly)
#if defined(__has_builtin)
#if __has_builtin(__builtin_amdgcn_tensor_load_to_lds) && __has_builtin(__builtin_amdgcn_s_wait_tensorcnt)
#define USE_TDM 1
#endif
#endif
#ifndef USE_TDM
#define USE_TDM 0
#endif

// ---------------- small prep kernels ----------------

// params[0..31]=d1 (t1-t0), params[32..63]=w (t0-2t1+t2), params[64]=S0 (sum t0)
__global__ void k_prep(const float* __restrict__ theta_log, float* __restrict__ params) {
    int k = threadIdx.x;
    __shared__ float t0s[32];
    if (k < 32) {
        float x0 = theta_log[k * 3 + 0];
        float x1 = theta_log[k * 3 + 1];
        float x2 = theta_log[k * 3 + 2];
        float t0 = -log1pf(__expf(-x0));   // log_sigmoid
        float t1 = -log1pf(__expf(-x1));
        float t2 = -log1pf(__expf(-x2));
        params[k]      = t1 - t0;
        params[32 + k] = t0 - 2.0f * t1 + t2;
        t0s[k] = t0;
    }
    __syncthreads();
    if (k == 0) {
        float s = 0.0f;
        for (int i = 0; i < 32; i++) s += t0s[i];
        params[64] = s;
    }
}

// softmax over 2048 seed_prob values, one block of 1024 threads
__global__ __launch_bounds__(1024) void k_seed(const float* __restrict__ sp, float* __restrict__ seed) {
    __shared__ float red[1024];
    int t = threadIdx.x;
    float a = sp[t], b = sp[t + 1024];
    red[t] = fmaxf(a, b);
    __syncthreads();
    for (int o = 512; o > 0; o >>= 1) {
        if (t < o) red[t] = fmaxf(red[t], red[t + o]);
        __syncthreads();
    }
    float mx = red[0];
    __syncthreads();
    float ea = __expf(a - mx), eb = __expf(b - mx);
    red[t] = ea + eb;
    __syncthreads();
    for (int o = 512; o > 0; o >>= 1) {
        if (t < o) red[t] += red[t + o];
        __syncthreads();
    }
    float inv = 1.0f / red[0];
    seed[t]        = ea * inv;
    seed[t + 1024] = eb * inv;
}

// pack c2a rows into bitmasks; u[i] = sum_k A[i,k]*(t1-t0)[k]
__global__ void k_bits(const int* __restrict__ c2a, const float* __restrict__ params,
                       unsigned* __restrict__ abits, float* __restrict__ u) {
    int i = blockIdx.x * blockDim.x + threadIdx.x;
    if (i >= NC_DIM) return;
    unsigned bits = 0;
    float uu = 0.0f;
    #pragma unroll
    for (int k = 0; k < 32; k++) {
        if (c2a[i * 32 + k]) { bits |= (1u << k); uu += params[k]; }
    }
    abits[i] = bits;
    u[i] = uu;
}

// E = exp(P) with unit diagonal (P symmetric), stored as bf16
__global__ void k_E(const unsigned* __restrict__ abits, const float* __restrict__ u,
                    const float* __restrict__ params, __bf16* __restrict__ Ebf) {
    __shared__ float wsh[32];
    __shared__ float S0sh;
    int t = threadIdx.y * 16 + threadIdx.x;
    if (t < 32) wsh[t] = params[32 + t];
    if (t == 32) S0sh = params[64];
    __syncthreads();
    int j = blockIdx.x * 16 + threadIdx.x;
    int i = blockIdx.y * 16 + threadIdx.y;
    unsigned both = abits[i] & abits[j];
    float p = S0sh + u[i] + u[j];
    while (both) {
        int k = __ffs(both) - 1;
        p += wsh[k];
        both &= both - 1;
    }
    float val = (i == j) ? 1.0f : __expf(p);
    Ebf[(size_t)i * NC_DIM + j] = (__bf16)val;
}

// Wbf[m,n] = mask ? seed[n] : 0 (bf16); rw[m] = exact f32 row sum; rowcnt[m] = #ones
__global__ __launch_bounds__(256) void k_W(const int* __restrict__ Ic, const float* __restrict__ seed,
                                           __bf16* __restrict__ Wbf, float* __restrict__ rw,
                                           float* __restrict__ rowcnt) {
    int m = blockIdx.x;
    int t = threadIdx.x;
    float s = 0.0f, cnt = 0.0f;
    for (int c = t; c < NC_DIM; c += 256) {
        int ic = Ic[(size_t)m * NC_DIM + c];
        float v = (ic == 1) ? seed[c] : 0.0f;
        Wbf[(size_t)m * NC_DIM + c] = (__bf16)v;
        s += v;
        if (ic == 1) cnt += 1.0f;
    }
    __shared__ float red[256];
    red[t] = s; __syncthreads();
    for (int o = 128; o > 0; o >>= 1) { if (t < o) red[t] += red[t + o]; __syncthreads(); }
    if (t == 0) rw[m] = red[0];
    __syncthreads();
    red[t] = cnt; __syncthreads();
    for (int o = 128; o > 0; o >>= 1) { if (t < o) red[t] += red[t + o]; __syncthreads(); }
    if (t == 0) rowcnt[m] = red[0];
}

// column sums of Ic
__global__ void k_colI(const int* __restrict__ Ic, float* __restrict__ colsumI) {
    int n = blockIdx.x * blockDim.x + threadIdx.x;
    if (n >= NC_DIM) return;
    float s = 0.0f;
    for (int m = 0; m < M_DIM; m++) s += (float)Ic[(size_t)m * NC_DIM + n];
    colsumI[n] = s;
}

// ---------------- main WMMA GEMM + fused epilogue ----------------
// exp_I = diag(1/rw) * Wbf @ Ebf ; accumulates loss, row sums, col sums.
// Double-buffered LDS staging; TDM (tensor_load_to_lds) prefetch when available.

#define LDS_STRIDE 40  // bf16 elems per LDS row: 80B = 64B data + 16B pad (conflict-free, 16B aligned)

#if USE_TDM
// One TDM descriptor loads a 128-row x 32-elem bf16 tile with 16B LDS row padding.
__device__ __forceinline__ void tdm_tile_load(const __bf16* gtile, unsigned lds_addr, int rows_total) {
    unsigned long long ga = (unsigned long long)(uintptr_t)gtile;
    v4u g0;
    g0.x = 1u;                                                  // count=1, user mode
    g0.y = lds_addr;                                            // LDS byte address
    g0.z = (unsigned)ga;                                        // global_addr[31:0]
    g0.w = (unsigned)((ga >> 32) & 0x01FFFFFFull) | (2u << 30); // global_addr[56:32] | type=2
    v8i g1;
    // data_size=2B (1<<16) | pad_enable (1<<20) | pad_interval=16DW (3<<22) | pad_amount=4DW (3<<25)
    g1[0] = (int)((1u << 16) | (1u << 20) | (3u << 22) | (3u << 25));
    g1[1] = (int)(((unsigned)NC_DIM & 0xFFFFu) << 16);          // tensor_dim0[15:0] @ bits[63:48]
    g1[2] = (int)((((unsigned)NC_DIM >> 16) & 0xFFFFu)          // tensor_dim0[31:16]
                  | (((unsigned)rows_total & 0xFFFFu) << 16));  // tensor_dim1[15:0]
    g1[3] = (int)(((((unsigned)rows_total) >> 16) & 0xFFFFu)    // tensor_dim1[31:16]
                  | (32u << 16));                               // tile_dim0 = 32 elems
    g1[4] = 128;                                                // tile_dim1 = 128 rows, tile_dim2 = 0
    g1[5] = NC_DIM;                                             // tensor_dim0_stride low
    g1[6] = 0;
    g1[7] = 0;
    v4i gz = {0, 0, 0, 0};
#if defined(__clang_major__) && (__clang_major__ >= 23)
    v8i gz8 = {0, 0, 0, 0, 0, 0, 0, 0};
    __builtin_amdgcn_tensor_load_to_lds(g0, g1, gz, gz, gz8, 0);
#else
    __builtin_amdgcn_tensor_load_to_lds(g0, g1, gz, gz, 0);
#endif
}
#endif

__global__ __launch_bounds__(256) void k_gemm(const __bf16* __restrict__ Wbf,
                                              const __bf16* __restrict__ Ebf,
                                              const float* __restrict__ rw,
                                              const int*  __restrict__ Ic,
                                              float* __restrict__ rowsumE,
                                              float* __restrict__ colsumE,
                                              float* __restrict__ loss) {
    __shared__ alignas(16) __bf16 As[2][128 * LDS_STRIDE];
    __shared__ alignas(16) __bf16 Bs[2][128 * LDS_STRIDE];
    __shared__ float dsRow[128];
    __shared__ float dsCol[128];
    __shared__ float dsLoss;

    const int tid = threadIdx.x;
    const int m0 = blockIdx.y * 128;
    const int n0 = blockIdx.x * 128;

    if (tid < 128) dsRow[tid] = 0.0f;
    else           dsCol[tid - 128] = 0.0f;
    if (tid == 0)  dsLoss = 0.0f;

    const int lane = tid & 31;
    const int wid  = tid >> 5;
    const int wm   = (wid & 3) * 32;   // wave row offset (2 m-tiles of 16)
    const int wn   = (wid >> 2) * 64;  // wave col offset (4 n-tiles of 16)
    const int hf   = lane >> 4;        // lane half
    const int l15  = lane & 15;
    const int kA   = hf ? 8 : 0;       // A frag: chunks at K {kA, kA+16}
    const int kB   = hf ? 8 : 0;       // B frag: chunks at K {kB, kB+16}

    v8f zero = {};
    v8f acc[2][4];
    #pragma unroll
    for (int mt = 0; mt < 2; mt++)
        #pragma unroll
        for (int nt = 0; nt < 4; nt++) acc[mt][nt] = zero;

#if USE_TDM
    // ---- prologue: DMA tile j0=0 into buffer 0 ----
    if (wid == 0)      tdm_tile_load(Wbf + (size_t)m0 * NC_DIM, (unsigned)(uintptr_t)&As[0][0], M_DIM);
    else if (wid == 1) tdm_tile_load(Ebf + (size_t)n0 * NC_DIM, (unsigned)(uintptr_t)&Bs[0][0], NC_DIM);
    if (wid < 2) __builtin_amdgcn_s_wait_tensorcnt(0);
    __syncthreads();
#else
    // ---- prologue: cooperative load of tile j0=0 into buffer 0 ----
    #pragma unroll
    for (int c = tid; c < 512; c += 256) {
        int row = c >> 2, ch = c & 3;
        *(uint4*)(&As[0][row * LDS_STRIDE + ch * 8]) =
            *(const uint4*)(Wbf + (size_t)(m0 + row) * NC_DIM + ch * 8);
        *(uint4*)(&Bs[0][row * LDS_STRIDE + ch * 8]) =
            *(const uint4*)(Ebf + (size_t)(n0 + row) * NC_DIM + ch * 8);
    }
    __syncthreads();
#endif

    int cur = 0;
    for (int j0 = 0; j0 < NC_DIM; j0 += 32) {
        const int nxt = cur ^ 1;
        const bool has_next = (j0 + 32) < NC_DIM;

#if USE_TDM
        // ---- async DMA prefetch of next tile into the other buffer ----
        if (has_next) {
            if (wid == 0)
                tdm_tile_load(Wbf + (size_t)m0 * NC_DIM + (j0 + 32),
                              (unsigned)(uintptr_t)&As[nxt][0], M_DIM);
            else if (wid == 1)
                tdm_tile_load(Ebf + (size_t)n0 * NC_DIM + (j0 + 32),
                              (unsigned)(uintptr_t)&Bs[nxt][0], NC_DIM);
        }
#else
        // ---- issue global loads of next tile into registers (overlap with WMMA) ----
        uint4 ra[2], rb[2];
        if (has_next) {
            #pragma unroll
            for (int q = 0; q < 2; q++) {
                int c = tid + q * 256;
                int row = c >> 2, ch = c & 3;
                ra[q] = *(const uint4*)(Wbf + (size_t)(m0 + row) * NC_DIM + (j0 + 32) + ch * 8);
                rb[q] = *(const uint4*)(Ebf + (size_t)(n0 + row) * NC_DIM + (j0 + 32) + ch * 8);
            }
        }
#endif

        // ---- fragments + 8 WMMAs from current buffer ----
        v16bf afrag[2], bfrag[4];
        #pragma unroll
        for (int mt = 0; mt < 2; mt++) {
            const __bf16* p = &As[cur][(wm + mt * 16 + l15) * LDS_STRIDE + kA];
            union { uint4 q[2]; v16bf v; } uu;
            uu.q[0] = *(const uint4*)p;
            uu.q[1] = *(const uint4*)(p + 16);
            afrag[mt] = uu.v;
        }
        #pragma unroll
        for (int nt = 0; nt < 4; nt++) {
            const __bf16* p = &Bs[cur][(wn + nt * 16 + l15) * LDS_STRIDE + kB];
            union { uint4 q[2]; v16bf v; } uu;
            uu.q[0] = *(const uint4*)p;
            uu.q[1] = *(const uint4*)(p + 16);
            bfrag[nt] = uu.v;
        }
        #pragma unroll
        for (int mt = 0; mt < 2; mt++)
            #pragma unroll
            for (int nt = 0; nt < 4; nt++)
                acc[mt][nt] = __builtin_amdgcn_wmma_f32_16x16x32_bf16(
                    false, afrag[mt], false, bfrag[nt], (short)0, acc[mt][nt], false, false);

#if USE_TDM
        if (has_next && wid < 2) __builtin_amdgcn_s_wait_tensorcnt(0);
#else
        if (has_next) {
            #pragma unroll
            for (int q = 0; q < 2; q++) {
                int c = tid + q * 256;
                int row = c >> 2, ch = c & 3;
                *(uint4*)(&As[nxt][row * LDS_STRIDE + ch * 8]) = ra[q];
                *(uint4*)(&Bs[nxt][row * LDS_STRIDE + ch * 8]) = rb[q];
            }
        }
#endif
        __syncthreads();
        cur = nxt;
    }

    // fused epilogue: v = acc / rw[m]; loss terms + row/col sums
    float localLoss = 0.0f;
    #pragma unroll
    for (int mt = 0; mt < 2; mt++) {
        #pragma unroll
        for (int r = 0; r < 8; r++) {
            int ml = wm + mt * 16 + r + hf * 8;
            int m  = m0 + ml;
            float inv = 1.0f / rw[m];
            #pragma unroll
            for (int nt = 0; nt < 4; nt++) {
                int nl = wn + nt * 16 + l15;
                int n  = n0 + nl;
                float v = acc[mt][nt][r] * inv;
                int ic = Ic[(size_t)m * NC_DIM + n];
                localLoss += (ic == 1) ? -logf(v) : -log1pf(-v);
                atomicAdd(&dsRow[ml], v);
                atomicAdd(&dsCol[nl], v);
            }
        }
    }
    atomicAdd(&dsLoss, localLoss);
    __syncthreads();
    if (tid < 128) atomicAdd(&rowsumE[m0 + tid], dsRow[tid]);
    else           atomicAdd(&colsumE[n0 + tid - 128], dsCol[tid - 128]);
    if (tid == 0)  atomicAdd(loss, dsLoss);
}

// ---------------- sort + diff ----------------

// descending bitonic sort of n (power of 2, <= 8192) floats, single block
__global__ __launch_bounds__(1024) void ksort(float* __restrict__ data, int n) {
    __shared__ float buf[8192];
    for (int i = threadIdx.x; i < n; i += 1024) buf[i] = data[i];
    __syncthreads();
    for (int k = 2; k <= n; k <<= 1) {
        for (int j = k >> 1; j > 0; j >>= 1) {
            for (int t = threadIdx.x; t < n; t += 1024) {
                int ixj = t ^ j;
                if (ixj > t) {
                    float a = buf[t], b = buf[ixj];
                    bool desc = ((t & k) == 0);
                    if (desc ? (a < b) : (a > b)) { buf[t] = b; buf[ixj] = a; }
                }
            }
            __syncthreads();
        }
    }
    for (int i = threadIdx.x; i < n; i += 1024) data[i] = buf[i];
}

__global__ __launch_bounds__(1024) void kdiff(const float* __restrict__ a, const float* __restrict__ b,
                                              int n, float w, float* __restrict__ out) {
    __shared__ float red[1024];
    int t = threadIdx.x;
    float s = 0.0f;
    for (int i = t; i < n; i += 1024) { float d = a[i] - b[i]; s += d * d; }
    red[t] = s; __syncthreads();
    for (int o = 512; o > 0; o >>= 1) { if (t < o) red[t] += red[t + o]; __syncthreads(); }
    if (t == 0) atomicAdd(out, w * red[0] / (float)n);
}

// ---------------- launch ----------------

extern "C" void kernel_launch(void* const* d_in, const int* in_sizes, int n_in,
                              void* d_out, int out_size, void* d_ws, size_t ws_size,
                              hipStream_t stream) {
    const float* theta_log = (const float*)d_in[0];
    const float* seed_prob = (const float*)d_in[1];
    const int*   Ic        = (const int*)d_in[2];
    const int*   c2a       = (const int*)d_in[3];
    float* out = (float*)d_out;
    char* ws = (char*)d_ws;

    float*    params  = (float*)(ws + 0);        // 65 floats
    float*    seed    = (float*)(ws + 512);      // 2048
    float*    u       = (float*)(ws + 8704);     // 2048
    unsigned* abits   = (unsigned*)(ws + 16896); // 2048
    float*    rw      = (float*)(ws + 25088);    // 8192
    float*    rowcntI = (float*)(ws + 57856);    // 8192
    float*    rowsumE = (float*)(ws + 90624);    // 8192
    float*    colsumI = (float*)(ws + 123392);   // 2048
    float*    colsumE = (float*)(ws + 131584);   // 2048
    __bf16*   Wbf     = (__bf16*)(ws + 139776);                              // 8192x2048 bf16
    __bf16*   Ebf     = (__bf16*)(ws + 139776 + (size_t)M_DIM * NC_DIM * 2); // 2048x2048 bf16

    hipMemsetAsync(rowsumE, 0, M_DIM * sizeof(float), stream);
    hipMemsetAsync(colsumE, 0, NC_DIM * sizeof(float), stream);
    hipMemsetAsync(out, 0, sizeof(float), stream);

    k_prep<<<1, 32, 0, stream>>>(theta_log, params);
    k_seed<<<1, 1024, 0, stream>>>(seed_prob, seed);
    k_bits<<<NC_DIM / 256, 256, 0, stream>>>(c2a, params, abits, u);
    k_E<<<dim3(NC_DIM / 16, NC_DIM / 16), dim3(16, 16), 0, stream>>>(abits, u, params, Ebf);
    k_W<<<M_DIM, 256, 0, stream>>>(Ic, seed, Wbf, rw, rowcntI);
    k_colI<<<NC_DIM / 256, 256, 0, stream>>>(Ic, colsumI);

    k_gemm<<<dim3(NC_DIM / 128, M_DIM / 128), 256, 0, stream>>>(Wbf, Ebf, rw, Ic,
                                                                rowsumE, colsumE, out);

    ksort<<<1, 1024, 0, stream>>>(colsumE, NC_DIM);
    ksort<<<1, 1024, 0, stream>>>(colsumI, NC_DIM);
    ksort<<<1, 1024, 0, stream>>>(rowsumE, M_DIM);
    ksort<<<1, 1024, 0, stream>>>(rowcntI, M_DIM);
    kdiff<<<1, 1024, 0, stream>>>(colsumE, colsumI, NC_DIM, 1.0f, out); // W_D
    kdiff<<<1, 1024, 0, stream>>>(rowsumE, rowcntI, M_DIM, 1.0f, out);  // W_S
}
